// MixtralSparseMoeBlock_21861383537459
// MI455X (gfx1250) — compile-verified, matrix operand order checked by reading, not run on hardware
//
#include <hip/hip_runtime.h>
#include <hip/hip_bf16.h>
#include <stdint.h>

#define T_TOKENS 16384
#define HIDDEN   1024
#define FFN      3584
#define NEXP     8

typedef __attribute__((ext_vector_type(16))) __bf16 bf16x16;
typedef __attribute__((ext_vector_type(8)))  float  floatx8;
typedef __attribute__((ext_vector_type(2)))  float  float2v;
typedef __attribute__((ext_vector_type(2)))  __bf16 bf16x2;

// ---------------------------------------------------------------------------
// helpers
// ---------------------------------------------------------------------------
__device__ __forceinline__ unsigned pack_bf16x2(float lo, float hi) {
  float2v f = { lo, hi };
  bf16x2 t = __builtin_convertvector(f, bf16x2);  // v_cvt_pk_bf16_f32
  unsigned r;
  __builtin_memcpy(&r, &t, 4);
  return r;
}

// gfx1250 async global->LDS copy, 16B per active lane, tracked by ASYNCcnt.
// (Low 32 bits of a flat pointer to __shared__ are the LDS byte offset.)
__device__ __forceinline__ void async_copy_b128(unsigned lds_byte_addr,
                                                const void* gptr) {
  asm volatile("global_load_async_to_lds_b128 %0, %1, off"
               :
               : "v"(lds_byte_addr),
                 "v"((unsigned long long)(uintptr_t)gptr)
               : "memory");
}

__device__ __forceinline__ void wait_async_le1() {
  asm volatile("s_wait_asynccnt 0x1" ::: "memory");
}
__device__ __forceinline__ void wait_async_le0() {
  asm volatile("s_wait_asynccnt 0x0" ::: "memory");
}

// Fragment loader: 16x32 bf16 tile (row-major, stride 32 elems) -> v16bf per
// CDNA5 ISA layout: lane = row%16 ; lanes 0-15 take K 0-7 & 16-23,
// lanes 16-31 take K 8-15 & 24-31 (two contiguous 16B -> 2x ds_load_b128).
// Also serves B fragments when the B tile is stored N-major (N x K).
__device__ __forceinline__ bf16x16 load_frag16x32(const __bf16* tile, int lane) {
  const int row  = lane & 15;
  const int half = lane >> 4;
  const __bf16* p = tile + row * 32 + half * 8;
  bf16x16 f;
#pragma unroll
  for (int i = 0; i < 8; ++i) { f[i] = p[i]; f[i + 8] = p[i + 16]; }
  return f;
}

// ---------------------------------------------------------------------------
// Router: one wave32 per token. logits = x @ gate_w^T + gate_b,
// softmax -> top2 -> renormalize.
// ---------------------------------------------------------------------------
__global__ __launch_bounds__(256) void moe_router(
    const float* __restrict__ x, const float* __restrict__ gw,
    const float* __restrict__ gb, int* __restrict__ topk_idx,
    float* __restrict__ topk_w) {
  const int wid  = threadIdx.x >> 5;
  const int lane = threadIdx.x & 31;
  const int t    = blockIdx.x * 8 + wid;
  if (t >= T_TOKENS) return;

  float acc[NEXP];
#pragma unroll
  for (int e = 0; e < NEXP; ++e) acc[e] = 0.f;
  const float* xr = x + (size_t)t * HIDDEN;
  for (int k = lane; k < HIDDEN; k += 32) {
    const float xv = xr[k];
#pragma unroll
    for (int e = 0; e < NEXP; ++e) acc[e] += xv * gw[e * HIDDEN + k];
  }
#pragma unroll
  for (int e = 0; e < NEXP; ++e) {
#pragma unroll
    for (int off = 16; off > 0; off >>= 1)
      acc[e] += __shfl_xor(acc[e], off, 32);
  }
  if (lane == 0) {
    float l[NEXP], m = -1e30f;
#pragma unroll
    for (int e = 0; e < NEXP; ++e) { l[e] = acc[e] + gb[e]; m = fmaxf(m, l[e]); }
    float p[NEXP];
#pragma unroll
    for (int e = 0; e < NEXP; ++e) p[e] = __expf(l[e] - m);
    int i0 = 0;
#pragma unroll
    for (int e = 1; e < NEXP; ++e) if (p[e] > p[i0]) i0 = e;
    int i1 = (i0 == 0) ? 1 : 0;
#pragma unroll
    for (int e = 0; e < NEXP; ++e) if (e != i0 && p[e] > p[i1]) i1 = e;
    const float s = p[i0] + p[i1];
    topk_idx[2 * t + 0] = i0;
    topk_idx[2 * t + 1] = i1;
    topk_w[2 * t + 0] = p[i0] / s;
    topk_w[2 * t + 1] = p[i1] / s;
  }
}

// ---------------------------------------------------------------------------
// Compaction: 8 waves, wave e compacts tokens that picked expert e
// (ballot prefix). Deterministic ordering, no atomics.
// ---------------------------------------------------------------------------
__global__ __launch_bounds__(256) void moe_compact(
    const int* __restrict__ topk_idx, const float* __restrict__ topk_w,
    int* __restrict__ row_token, float* __restrict__ row_wt,
    int* __restrict__ counts) {
  const int e    = threadIdx.x >> 5;
  const int lane = threadIdx.x & 31;
  if (e >= NEXP) return;
  int base = 0;
  for (int t0 = 0; t0 < T_TOKENS; t0 += 32) {
    const int t  = t0 + lane;
    const int i0 = topk_idx[2 * t + 0];
    const int i1 = topk_idx[2 * t + 1];
    const bool flag = (i0 == e) || (i1 == e);
    const float w = (i0 == e) ? topk_w[2 * t + 0] : topk_w[2 * t + 1];
    const unsigned mask = (unsigned)__ballot(flag);
    const int pos = base + __popc(mask & ((1u << lane) - 1u));
    if (flag) {
      row_token[e * T_TOKENS + pos] = t;
      row_wt[e * T_TOKENS + pos]    = w;
    }
    base += __popc(mask);
  }
  if (lane == 0) counts[e] = base;
}

// ---------------------------------------------------------------------------
// x (fp32) -> Xb (bf16)
// ---------------------------------------------------------------------------
__global__ __launch_bounds__(256) void moe_cvt_bf16(
    const float* __restrict__ x, __bf16* __restrict__ xb) {
  const size_t i4 = (size_t)blockIdx.x * blockDim.x + threadIdx.x;
  const float4 v = ((const float4*)x)[i4];
  unsigned o[2] = { pack_bf16x2(v.x, v.y), pack_bf16x2(v.z, v.w) };
  *((uint2*)(xb + i4 * 4)) = *((uint2*)o);
}

// ---------------------------------------------------------------------------
// GEMM1 (per expert): C1 = gather(Xb) @ W1e, C3 = gather(Xb) @ W3e,
// H = silu(C1)*C3 (bf16). BM=64, BN=128, 8 waves (2x4), wave tile 32x32.
// A tile: double-buffered async global->LDS copy. W tiles: register-pipelined
// fp32 loads through running pointers (strength-reduced addressing),
// pair-packed bf16 stores (N-major) into LDS.
// ---------------------------------------------------------------------------
__global__ __launch_bounds__(256) void moe_gemm1(
    const __bf16* __restrict__ Xb, const float* __restrict__ w1,
    const float* __restrict__ w3, const int* __restrict__ row_token,
    const int* __restrict__ counts, __bf16* __restrict__ Hbuf, int e) {
  __shared__ __bf16 Xt[2][64 * 32];
  __shared__ __bf16 W1t[128 * 32];
  __shared__ __bf16 W3t[128 * 32];
  __shared__ int    toks[64];

  const int cnt   = counts[e];
  const int mBase = blockIdx.y * 64;
  if (mBase >= cnt) return;
  const int nBase = blockIdx.x * 128;
  const int tid   = threadIdx.x;
  const int lane  = tid & 31;
  const int wid   = tid >> 5;
  const int waveM = (wid >> 2) * 32;
  const int waveN = (wid & 3) * 32;

  const float* w1e = w1 + (size_t)e * HIDDEN * FFN + nBase;
  const float* w3e = w3 + (size_t)e * HIDDEN * FFN + nBase;

  if (tid < 64)
    toks[tid] = (mBase + tid < cnt) ? row_token[e * T_TOKENS + mBase + tid] : 0;
  __syncthreads();

  // A-copy lane assignment; running source pointer (advance 32 elems / iter)
  const int arow = tid >> 2, aseg = tid & 3;
  const __bf16* asrc = Xb + (size_t)toks[arow] * HIDDEN + aseg * 8;
  const unsigned aldsOff = (unsigned)((arow * 32 + aseg * 8) * 2);
  const unsigned xtBase[2] = { (unsigned)(uintptr_t)&Xt[0][0],
                               (unsigned)(uintptr_t)&Xt[1][0] };

  // W staging: 2 units/thread/matrix; running pointers, advance 32 rows/iter
  int kp[2], n4[2];
  const float* wp1[2];
  const float* wp3[2];
#pragma unroll
  for (int u = 0; u < 2; ++u) {
    const int q = tid + u * 256;
    kp[u] = q >> 5;
    n4[u] = q & 31;
    wp1[u] = w1e + (size_t)(kp[u] * 2) * FFN + n4[u] * 4;
    wp3[u] = w3e + (size_t)(kp[u] * 2) * FFN + n4[u] * 4;
  }

  float4 w1r[2][2], w3r[2][2];
  auto loadW = [&]() {  // load current tile, advance pointers by K-step
#pragma unroll
    for (int u = 0; u < 2; ++u) {
      w1r[u][0] = *(const float4*)(wp1[u]);
      w1r[u][1] = *(const float4*)(wp1[u] + FFN);
      w3r[u][0] = *(const float4*)(wp3[u]);
      w3r[u][1] = *(const float4*)(wp3[u] + FFN);
      wp1[u] += 32 * FFN;
      wp3[u] += 32 * FFN;
    }
  };
  auto storeW = [&]() {
#pragma unroll
    for (int u = 0; u < 2; ++u) {
      const int k = kp[u] * 2;
#pragma unroll
      for (int j = 0; j < 4; ++j) {
        const int n = n4[u] * 4 + j;
        *(unsigned*)(&W1t[n * 32 + k]) =
            pack_bf16x2((&w1r[u][0].x)[j], (&w1r[u][1].x)[j]);
        *(unsigned*)(&W3t[n * 32 + k]) =
            pack_bf16x2((&w3r[u][0].x)[j], (&w3r[u][1].x)[j]);
      }
    }
  };

  floatx8 c1[2][2], c3[2][2];
#pragma unroll
  for (int a = 0; a < 2; ++a)
#pragma unroll
    for (int b = 0; b < 2; ++b) { c1[a][b] = (floatx8)0.f; c3[a][b] = (floatx8)0.f; }

  async_copy_b128(xtBase[0] + aldsOff, asrc);   // A(0)
  asrc += 32;
  loadW();                                       // W(0)

  const int KITER = HIDDEN / 32;
  for (int kk = 0; kk < KITER; ++kk) {
    const int cur = kk & 1;
    if (kk + 1 < KITER) {
      async_copy_b128(xtBase[cur ^ 1] + aldsOff, asrc);  // A(kk+1)
      asrc += 32;
    }
    storeW();
    if (kk + 1 < KITER) wait_async_le1();  // A(kk) landed (A(kk+1) in flight)
    else                wait_async_le0();
    __syncthreads();                       // A(kk) + W(kk) visible to all waves
    if (kk + 1 < KITER) loadW();           // overlap next W globals with WMMA

    const __bf16* xcur = &Xt[cur][0];
    bf16x16 aF[2], b1F[2], b3F[2];
#pragma unroll
    for (int mt = 0; mt < 2; ++mt)
      aF[mt] = load_frag16x32(xcur + (waveM + mt * 16) * 32, lane);
#pragma unroll
    for (int nt = 0; nt < 2; ++nt) {
      b1F[nt] = load_frag16x32(&W1t[(waveN + nt * 16) * 32], lane);
      b3F[nt] = load_frag16x32(&W3t[(waveN + nt * 16) * 32], lane);
    }
#pragma unroll
    for (int mt = 0; mt < 2; ++mt)
#pragma unroll
      for (int nt = 0; nt < 2; ++nt) {
        c1[mt][nt] = __builtin_amdgcn_wmma_f32_16x16x32_bf16(
            false, aF[mt], false, b1F[nt], (short)0, c1[mt][nt], false, false);
        c3[mt][nt] = __builtin_amdgcn_wmma_f32_16x16x32_bf16(
            false, aF[mt], false, b3F[nt], (short)0, c3[mt][nt], false, false);
      }
    __syncthreads();                       // W LDS free for next store
  }

  // epilogue: SwiGLU, store bf16 H. C layout: n = lane%16, m = (lane/16)*8+v.
  const int nIn = lane & 15, halfSel = lane >> 4;
#pragma unroll
  for (int mt = 0; mt < 2; ++mt)
#pragma unroll
    for (int nt = 0; nt < 2; ++nt) {
#pragma unroll
      for (int v = 0; v < 8; ++v) {
        const int m   = halfSel * 8 + v;
        const int row = mBase + waveM + mt * 16 + m;
        if (row < cnt) {
          const float a  = c1[mt][nt][v];
          const float g  = a * __builtin_amdgcn_rcpf(1.f + __expf(-a));  // silu
          const float hv = g * c3[mt][nt][v];
          Hbuf[(size_t)row * FFN + nBase + waveN + nt * 16 + nIn] = (__bf16)hv;
        }
      }
    }
}

// ---------------------------------------------------------------------------
// GEMM2 (per expert): out[token] += wt * (H_row @ W2e). Same pipeline, K=FFN.
// ---------------------------------------------------------------------------
__global__ __launch_bounds__(256) void moe_gemm2(
    const __bf16* __restrict__ Hbuf, const float* __restrict__ w2,
    const int* __restrict__ row_token, const float* __restrict__ row_wt,
    const int* __restrict__ counts, float* __restrict__ out, int e) {
  __shared__ __bf16 At[2][64 * 32];
  __shared__ __bf16 Bt[128 * 32];
  __shared__ int    toks[64];
  __shared__ float  wts[64];

  const int cnt   = counts[e];
  const int mBase = blockIdx.y * 64;
  if (mBase >= cnt) return;
  const int nBase = blockIdx.x * 128;
  const int tid   = threadIdx.x;
  const int lane  = tid & 31;
  const int wid   = tid >> 5;
  const int waveM = (wid >> 2) * 32;
  const int waveN = (wid & 3) * 32;

  const float* w2e = w2 + (size_t)e * FFN * HIDDEN + nBase;

  if (tid < 64) {
    const bool ok = (mBase + tid < cnt);
    toks[tid] = ok ? row_token[e * T_TOKENS + mBase + tid] : 0;
    wts[tid]  = ok ? row_wt[e * T_TOKENS + mBase + tid] : 0.f;
  }
  __syncthreads();

  const int arow = tid >> 2, aseg = tid & 3;
  const __bf16* asrc = Hbuf + (size_t)(mBase + arow) * FFN + aseg * 8;
  const unsigned aldsOff = (unsigned)((arow * 32 + aseg * 8) * 2);
  const unsigned atBase[2] = { (unsigned)(uintptr_t)&At[0][0],
                               (unsigned)(uintptr_t)&At[1][0] };

  int kp[2], n4[2];
  const float* wp2[2];
#pragma unroll
  for (int u = 0; u < 2; ++u) {
    const int q = tid + u * 256;
    kp[u] = q >> 5;
    n4[u] = q & 31;
    wp2[u] = w2e + (size_t)(kp[u] * 2) * HIDDEN + n4[u] * 4;
  }

  float4 w2r[2][2];
  auto loadW = [&]() {
#pragma unroll
    for (int u = 0; u < 2; ++u) {
      w2r[u][0] = *(const float4*)(wp2[u]);
      w2r[u][1] = *(const float4*)(wp2[u] + HIDDEN);
      wp2[u] += 32 * HIDDEN;
    }
  };
  auto storeW = [&]() {
#pragma unroll
    for (int u = 0; u < 2; ++u) {
      const int k = kp[u] * 2;
#pragma unroll
      for (int j = 0; j < 4; ++j) {
        const int n = n4[u] * 4 + j;
        *(unsigned*)(&Bt[n * 32 + k]) =
            pack_bf16x2((&w2r[u][0].x)[j], (&w2r[u][1].x)[j]);
      }
    }
  };

  floatx8 c[2][2];
#pragma unroll
  for (int a = 0; a < 2; ++a)
#pragma unroll
    for (int b = 0; b < 2; ++b) c[a][b] = (floatx8)0.f;

  async_copy_b128(atBase[0] + aldsOff, asrc);
  asrc += 32;
  loadW();

  const int KITER = FFN / 32;
  for (int kk = 0; kk < KITER; ++kk) {
    const int cur = kk & 1;
    if (kk + 1 < KITER) {
      async_copy_b128(atBase[cur ^ 1] + aldsOff, asrc);
      asrc += 32;
    }
    storeW();
    if (kk + 1 < KITER) wait_async_le1();
    else                wait_async_le0();
    __syncthreads();
    if (kk + 1 < KITER) loadW();

    const __bf16* acur = &At[cur][0];
    bf16x16 aF[2], bF[2];
#pragma unroll
    for (int mt = 0; mt < 2; ++mt)
      aF[mt] = load_frag16x32(acur + (waveM + mt * 16) * 32, lane);
#pragma unroll
    for (int nt = 0; nt < 2; ++nt)
      bF[nt] = load_frag16x32(&Bt[(waveN + nt * 16) * 32], lane);
#pragma unroll
    for (int mt = 0; mt < 2; ++mt)
#pragma unroll
      for (int nt = 0; nt < 2; ++nt)
        c[mt][nt] = __builtin_amdgcn_wmma_f32_16x16x32_bf16(
            false, aF[mt], false, bF[nt], (short)0, c[mt][nt], false, false);
    __syncthreads();
  }

  const int nIn = lane & 15, halfSel = lane >> 4;
#pragma unroll
  for (int mt = 0; mt < 2; ++mt)
#pragma unroll
    for (int nt = 0; nt < 2; ++nt) {
#pragma unroll
      for (int v = 0; v < 8; ++v) {
        const int m        = halfSel * 8 + v;
        const int rowLocal = waveM + mt * 16 + m;
        const int row      = mBase + rowLocal;
        if (row < cnt) {
          const int tok = toks[rowLocal];
          const float w = wts[rowLocal];
          float* o = out + (size_t)tok * HIDDEN + nBase + waveN + nt * 16 + nIn;
          *o += w * c[mt][nt][v];
        }
      }
    }
}

// ---------------------------------------------------------------------------
extern "C" void kernel_launch(void* const* d_in, const int* in_sizes, int n_in,
                              void* d_out, int out_size, void* d_ws,
                              size_t ws_size, hipStream_t stream) {
  const float* x  = (const float*)d_in[0];
  const float* gw = (const float*)d_in[1];
  const float* gb = (const float*)d_in[2];
  const float* w1 = (const float*)d_in[3];
  const float* w3 = (const float*)d_in[4];
  const float* w2 = (const float*)d_in[5];
  float* out = (float*)d_out;

  char* ws = (char*)d_ws;
  size_t off = 0;
  auto alloc = [&](size_t bytes) -> void* {
    void* p = ws + off;
    off += (bytes + 255) & ~(size_t)255;
    return p;
  };
  int*    topk_idx  = (int*)alloc((size_t)2 * T_TOKENS * sizeof(int));
  float*  topk_w    = (float*)alloc((size_t)2 * T_TOKENS * sizeof(float));
  int*    counts    = (int*)alloc(64);
  int*    row_token = (int*)alloc((size_t)NEXP * T_TOKENS * sizeof(int));
  float*  row_wt    = (float*)alloc((size_t)NEXP * T_TOKENS * sizeof(float));
  __bf16* Xb        = (__bf16*)alloc((size_t)T_TOKENS * HIDDEN * 2);
  __bf16* Hbuf      = (__bf16*)alloc((size_t)T_TOKENS * FFN * 2);
  (void)ws_size; (void)in_sizes; (void)n_in; (void)out_size;

  hipMemsetAsync(d_out, 0, (size_t)T_TOKENS * HIDDEN * sizeof(float), stream);

  moe_router<<<T_TOKENS / 8, 256, 0, stream>>>(x, gw, gb, topk_idx, topk_w);
  moe_compact<<<1, 256, 0, stream>>>(topk_idx, topk_w, row_token, row_wt, counts);
  moe_cvt_bf16<<<(T_TOKENS * HIDDEN / 4) / 256, 256, 0, stream>>>(x, Xb);

  for (int e = 0; e < NEXP; ++e) {
    moe_gemm1<<<dim3(FFN / 128, T_TOKENS / 64), 256, 0, stream>>>(
        Xb, w1, w3, row_token, counts, Hbuf, e);
    moe_gemm2<<<dim3(HIDDEN / 128, T_TOKENS / 64), 256, 0, stream>>>(
        Hbuf, w2, row_token, row_wt, counts, out, e);
  }
}
